// GRU_6579889898095
// MI455X (gfx1250) — matrix operand order repaired and checked by reading, not compiled, not run on hardware
//
#include <hip/hip_runtime.h>

#define B_  64
#define T_  512
#define IN_ 128
#define H_  512
#define NWG 16

typedef __bf16 bf16;
typedef __attribute__((ext_vector_type(8)))  bf16  v8bf;
typedef __attribute__((ext_vector_type(16))) bf16  v16bf;
typedef __attribute__((ext_vector_type(8)))  float v8f;

// ---------------------------------------------------------------------------
// WMMA helper: D = A(16x32 bf16) * B(32x16 bf16) + C(16x16 f32)
// ---------------------------------------------------------------------------
__device__ __forceinline__ v8f wmma_bf16(v16bf a, v16bf b, v8f c) {
  return __builtin_amdgcn_wmma_f32_16x16x32_bf16(
      /*neg_a=*/false, a, /*neg_b=*/false, b,
      /*c_mod=*/(short)0, c, /*reuse_a=*/false, /*reuse_b=*/false);
}

// A-matrix 16x32 bf16 fragment (ISA 7.12.2): lanes 0-15 -> M=lane, lanes 16-31
// -> M=lane-16 with K offset 8. Elements 0-7: K = kh+0..7, elements 8-15:
// K = 16+kh+0..7 (kh = 0 or 8). Two 16B loads per lane.
__device__ __forceinline__ v16bf load_a_frag(const bf16* base, size_t ld,
                                             int row0, int k0) {
  const int lane = threadIdx.x & 31;
  const int m    = lane & 15;
  const int kh   = (lane & 16) >> 1;        // 0 | 8
  const bf16* p  = base + (size_t)(row0 + m) * ld + (size_t)(k0 + kh);
  union { v16bf v; v8bf h[2]; } u;
  u.h[0] = *(const v8bf*)(p);
  u.h[1] = *(const v8bf*)(p + 16);
  return u.v;
}

// B-matrix 32x16 bf16 fragment: lanes hold N=lane&15; elements 0-7: K = kh+0..7,
// elements 8-15: K = 16+kh+0..7 (kh = 0 or 8), matching A. W is stored
// row-major [out_col, k], so this is two 16B loads per lane.
__device__ __forceinline__ v16bf load_b_frag(const bf16* Wb, size_t ld,
                                             int row0, int k0) {
  const int lane = threadIdx.x & 31;
  const int n    = lane & 15;
  const int kh   = (lane & 16) >> 1;        // 0 | 8
  const bf16* p  = Wb + (size_t)(row0 + n) * ld + (size_t)(k0 + kh);
  union { v16bf v; v8bf h[2]; } u;
  u.h[0] = *(const v8bf*)(p);
  u.h[1] = *(const v8bf*)(p + 16);
  return u.v;
}

__device__ __forceinline__ float sigmoidf_(float x) {
  return 1.0f / (1.0f + __expf(-x));
}

// Grid-wide sense-reversal barrier over NWG co-resident workgroups.
// bar[0] = arrival count, bar[1] = generation.
__device__ __forceinline__ void grid_barrier(unsigned* bar) {
  __syncthreads();
  if (threadIdx.x == 0) {
    __threadfence();                                   // release h writes
    volatile unsigned* gen = bar + 1;
    const unsigned g = *gen;
    const unsigned prev = atomicAdd(bar, 1u);
    if (prev == NWG - 1) {
      *(volatile unsigned*)bar = 0u;                   // reset for next round
      __threadfence();
      atomicAdd(bar + 1, 1u);                          // open the barrier
    } else {
      while (*gen == g) __builtin_amdgcn_s_sleep(1);
    }
    __threadfence();                                   // acquire h writes
  }
  __syncthreads();
}

// ---------------------------------------------------------------------------
// Persistent GRU layer. 16 WGs x 256 threads (8 waves). WG owns 32 hidden
// columns; wave owns (m_tile 0..3, c_tile 0..1) -> one 16x16 h tile.
// W_ih / W_hh column slices (96 rows x K each) live in LDS for the whole scan.
// ---------------------------------------------------------------------------
__global__ void __launch_bounds__(256, 1)
gru_layer(const bf16* __restrict__ Xin, int Din, size_t xRowStride,
          const bf16* __restrict__ Wih, const bf16* __restrict__ Whh,
          const float* __restrict__ bih, const float* __restrict__ bhh,
          float* __restrict__ hG, bf16* __restrict__ hbf,
          bf16* __restrict__ Hout, unsigned* __restrict__ bar) {
  extern __shared__ __align__(16) char smem_raw[];
  bf16* ldsIH = (bf16*)smem_raw;              // [96, Din]
  bf16* ldsHH = ldsIH + (size_t)96 * Din;     // [96, H_]

  const int tid   = threadIdx.x;
  const int lane  = tid & 31;
  const int wave  = tid >> 5;
  const int wgC0  = blockIdx.x * 32;          // first hidden column of this WG
  const int m0    = (wave & 3) * 16;          // batch-row tile
  const int c0loc = (wave >> 2) * 16;         // column tile within WG slice
  const int c0    = wgC0 + c0loc;

  // ---- Stage this WG's weight column slices into LDS (once) --------------
  {
    const int cprIH = Din >> 3;               // v8bf chunks per row
    for (int i = tid; i < 96 * cprIH; i += 256) {
      const int r = i / cprIH, ck = i - r * cprIH;
      const int gate = r >> 5, cloc = r & 31;
      ((v8bf*)ldsIH)[(size_t)r * cprIH + ck] =
          *(const v8bf*)(Wih + (size_t)(gate * H_ + wgC0 + cloc) * Din + ck * 8);
    }
    for (int i = tid; i < 96 * 64; i += 256) {
      const int r = i >> 6, ck = i & 63;
      const int gate = r >> 5, cloc = r & 31;
      ((v8bf*)ldsHH)[(size_t)r * 64 + ck] =
          *(const v8bf*)(Whh + (size_t)(gate * H_ + wgC0 + cloc) * H_ + ck * 8);
    }
  }

  // ---- h0 = 0 (each wave zeroes exactly its own tile) --------------------
  const int cN   = c0 + (lane & 15);          // this lane's hidden column
  const int rOff = (lane & 16) ? 8 : 0;       // C/D layout row offset
  for (int v = 0; v < 8; ++v) {
    const int b = m0 + rOff + v;
    hG[(size_t)b * H_ + cN]  = 0.0f;
    hbf[(size_t)b * H_ + cN] = (bf16)0.0f;
  }
  __syncthreads();
  grid_barrier(bar);

  // Biases are per-column -> hoist out of the time loop.
  const float bR  = bih[cN] + bhh[cN];
  const float bZ  = bih[H_ + cN] + bhh[H_ + cN];
  const float bXN = bih[2 * H_ + cN];
  const float bHN = bhh[2 * H_ + cN];

  for (int t = 0; t < T_; ++t) {
    v8f accR  = {0, 0, 0, 0, 0, 0, 0, 0};
    v8f accZ  = accR, accXN = accR, accHN = accR;

    const bf16* xbase = Xin + (size_t)t * Din;
    __builtin_prefetch(xbase + Din + (size_t)m0 * xRowStride, 0, 0);

    // x-projection: accR/accZ/accXN += x_t * W_ih^T (slice from LDS)
    for (int k0 = 0; k0 < Din; k0 += 32) {
      const v16bf a  = load_a_frag(xbase, xRowStride, m0, k0);
      const v16bf br = load_b_frag(ldsIH, Din, 0 * 32 + c0loc, k0);
      const v16bf bz = load_b_frag(ldsIH, Din, 1 * 32 + c0loc, k0);
      const v16bf bn = load_b_frag(ldsIH, Din, 2 * 32 + c0loc, k0);
      accR  = wmma_bf16(a, br, accR);
      accZ  = wmma_bf16(a, bz, accZ);
      accXN = wmma_bf16(a, bn, accXN);
    }
    // h-projection: accR/accZ/accHN += h_{t-1} * W_hh^T (slice from LDS)
    for (int k0 = 0; k0 < H_; k0 += 32) {
      const v16bf a  = load_a_frag(hbf, H_, m0, k0);
      const v16bf br = load_b_frag(ldsHH, H_, 0 * 32 + c0loc, k0);
      const v16bf bz = load_b_frag(ldsHH, H_, 1 * 32 + c0loc, k0);
      const v16bf bn = load_b_frag(ldsHH, H_, 2 * 32 + c0loc, k0);
      accR  = wmma_bf16(a, br, accR);
      accZ  = wmma_bf16(a, bz, accZ);
      accHN = wmma_bf16(a, bn, accHN);
    }

    // Gate math fully in registers (tile ownership is exclusive per lane).
#pragma unroll
    for (int v = 0; v < 8; ++v) {
      const int b   = m0 + rOff + v;
      const float r = sigmoidf_(accR[v] + bR);
      const float z = sigmoidf_(accZ[v] + bZ);
      const float n = tanhf(accXN[v] + bXN + r * (accHN[v] + bHN));
      const float hp = hG[(size_t)b * H_ + cN];
      const float hv = (1.0f - z) * n + z * hp;
      hG[(size_t)b * H_ + cN]  = hv;
      hbf[(size_t)b * H_ + cN] = (bf16)hv;
      if (Hout) Hout[((size_t)b * T_ + t) * H_ + cN] = (bf16)hv;
    }
    grid_barrier(bar);   // h_t visible device-wide before step t+1
  }
}

// ---------------------------------------------------------------------------
// Support kernels
// ---------------------------------------------------------------------------
__global__ void cvt_bf16_kernel(const float* __restrict__ src,
                                bf16* __restrict__ dst, int n) {
  const int i = blockIdx.x * blockDim.x + threadIdx.x;
  if (i < n) dst[i] = (bf16)src[i];
}

__global__ void zero_u32_kernel(unsigned* __restrict__ p, int n) {
  const int i = blockIdx.x * blockDim.x + threadIdx.x;
  if (i < n) p[i] = 0u;
}

__global__ void fc_kernel(const float* __restrict__ h, const float* __restrict__ W,
                          const float* __restrict__ b, float* __restrict__ out) {
  __shared__ float red[256];
  const int bi = blockIdx.x;
  float s = 0.0f;
  for (int i = threadIdx.x; i < H_; i += 256) s += h[(size_t)bi * H_ + i] * W[i];
  red[threadIdx.x] = s;
  __syncthreads();
  for (int off = 128; off > 0; off >>= 1) {
    if ((int)threadIdx.x < off) red[threadIdx.x] += red[threadIdx.x + off];
    __syncthreads();
  }
  if (threadIdx.x == 0) out[bi] = red[0] + b[0];
}

// ---------------------------------------------------------------------------
// Host-side launcher
// ---------------------------------------------------------------------------
extern "C" void kernel_launch(void* const* d_in, const int* in_sizes, int n_in,
                              void* d_out, int out_size, void* d_ws, size_t ws_size,
                              hipStream_t stream) {
  const float* X    = (const float*)d_in[0];
  const float* Wih0 = (const float*)d_in[1];
  const float* Whh0 = (const float*)d_in[2];
  const float* bih0 = (const float*)d_in[3];
  const float* bhh0 = (const float*)d_in[4];
  const float* Wih1 = (const float*)d_in[5];
  const float* Whh1 = (const float*)d_in[6];
  const float* bih1 = (const float*)d_in[7];
  const float* bhh1 = (const float*)d_in[8];
  const float* fcW  = (const float*)d_in[9];
  const float* fcb  = (const float*)d_in[10];

  char* w = (char*)d_ws;
  auto carve = [&](size_t bytes) -> char* {
    char* p = w;
    w += (bytes + 255) & ~(size_t)255;
    return p;
  };
  bf16*  Xbf   = (bf16*)carve((size_t)B_ * T_ * IN_ * 2);  // layer0 input, bf16
  bf16*  H0    = (bf16*)carve((size_t)B_ * T_ * H_ * 2);   // layer0 output seq
  bf16*  Wih0b = (bf16*)carve((size_t)3 * H_ * IN_ * 2);
  bf16*  Whh0b = (bf16*)carve((size_t)3 * H_ * H_ * 2);
  bf16*  Wih1b = (bf16*)carve((size_t)3 * H_ * H_ * 2);
  bf16*  Whh1b = (bf16*)carve((size_t)3 * H_ * H_ * 2);
  float* hG    = (float*)carve((size_t)B_ * H_ * 4);       // fp32 hidden state
  bf16*  hbf   = (bf16*)carve((size_t)B_ * H_ * 2);        // bf16 mirror (A frags)
  unsigned* bars = (unsigned*)carve(512);                  // 2 barrier pairs

  auto cvt = [&](const float* s, bf16* d, int n) {
    cvt_bf16_kernel<<<(n + 255) / 256, 256, 0, stream>>>(s, d, n);
  };
  cvt(X, Xbf, B_ * T_ * IN_);
  cvt(Wih0, Wih0b, 3 * H_ * IN_);
  cvt(Whh0, Whh0b, 3 * H_ * H_);
  cvt(Wih1, Wih1b, 3 * H_ * H_);
  cvt(Whh1, Whh1b, 3 * H_ * H_);
  zero_u32_kernel<<<1, 128, 0, stream>>>(bars, 128);

  const size_t smem0 = (size_t)96 * IN_ * 2 + (size_t)96 * H_ * 2;  // 120 KB
  const size_t smem1 = (size_t)96 * H_ * 2 + (size_t)96 * H_ * 2;   // 192 KB
  hipFuncSetAttribute((const void*)gru_layer,
                      hipFuncAttributeMaxDynamicSharedMemorySize, (int)smem1);

  gru_layer<<<NWG, 256, smem0, stream>>>(Xbf, IN_, (size_t)T_ * IN_,
                                         Wih0b, Whh0b, bih0, bhh0,
                                         hG, hbf, H0, bars);
  gru_layer<<<NWG, 256, smem1, stream>>>(H0, H_, (size_t)T_ * H_,
                                         Wih1b, Whh1b, bih1, bhh1,
                                         hG, hbf, nullptr, bars + 64);
  fc_kernel<<<B_, 256, 0, stream>>>(hG, fcW, fcb, (float*)d_out);

  (void)in_sizes; (void)n_in; (void)out_size; (void)ws_size;
}
